// Lstm_Model_75359496175641
// MI455X (gfx1250) — compile-verified, matrix operand order checked by reading, not run on hardware
//
#include <hip/hip_runtime.h>

typedef __attribute__((ext_vector_type(16))) __bf16 v16bf;
typedef __attribute__((ext_vector_type(8)))  float  v8f;
typedef __attribute__((ext_vector_type(4)))  float  f32x4;

#define T_STEPS 2048
#define BSZ 128
#define ISZ 10
#define HSZ 512
#define NWG 16
#define NH_LOCAL 32            // hidden units owned per workgroup
#define NGATE 128              // 4 * NH_LOCAL gate rows per workgroup
#define THREADS 512            // 16 waves (wave32)

#define WSTRIDE 520            // padded bf16 row stride for W_hh / h tiles (bank-conflict pad)
#define XSTRIDE 40             // padded bf16 row stride for x / W_ih (K padded 10 -> 32)
#define GSTRIDE 132            // padded f32 row stride for gate buffer

// LDS carve (bytes); gate buffer (f32) aliases the h staging buffer (bf16)
#define OFF_W    0
#define OFF_H    (NGATE * WSTRIDE * 2)            // 133120
#define OFF_X    (OFF_H + BSZ * WSTRIDE * 2)      // 266240
#define OFF_WIH  (OFF_X + BSZ * XSTRIDE * 2)
#define OFF_BIAS (OFF_WIH + NGATE * XSTRIDE * 2)
#define OFF_C    (OFF_BIAS + 512)
#define OFF_XRAW (OFF_C + BSZ * NH_LOCAL * 4)     // raw f32 x_t staging (async dest)
#define LDS_BYTES (OFF_XRAW + BSZ * ISZ * 4)      // ~302 KB <= 320 KB WGP LDS

// 32-bit wave-relative LDS offset from a generic pointer (low 32 bits of the
// shared-aperture flat address are the LDS offset on gfx1250).
__device__ __forceinline__ unsigned lds_off(const void* p) {
  return (unsigned)(uintptr_t)p;
}

// Async global->LDS 16B copy (CDNA5 Tensor/Async path, tracked by ASYNCcnt).
// IOFFSET is added to BOTH the global and LDS addresses.
__device__ __forceinline__ void async_copy_b128(unsigned lds_byte_off,
                                                unsigned long long gaddr) {
  asm volatile("global_load_async_to_lds_b128 %0, %1, off"
               :: "v"(lds_byte_off), "v"(gaddr) : "memory");
}

__device__ __forceinline__ void wait_asynccnt0() {
#if __has_builtin(__builtin_amdgcn_s_wait_asynccnt)
  __builtin_amdgcn_s_wait_asynccnt(0);
#else
  asm volatile("s_wait_asynccnt 0" ::: "memory");
#endif
}

// Load one 16x32 bf16 WMMA fragment (A or B role; both mirror the same
// per-lane layout: lanes 0-15 -> K {0..7,16..23}, lanes 16-31 -> K {8..15,24..31}).
__device__ __forceinline__ v16bf load_frag(const __bf16* base, int row0, int stride,
                                           int k0, int lane) {
  const int half = lane >> 4;
  const int r = row0 + (lane & 15);
  const __bf16* p = base + r * stride + k0 + 8 * half;
  union { v16bf v; f32x4 f[2]; } u;
  u.f[0] = *(const f32x4*)(p);        // K +0..7  (of this half)
  u.f[1] = *(const f32x4*)(p + 16);   // K +16..23 (of this half)
  return u.v;
}

#define WMMA_BF16(A, B, C) \
  __builtin_amdgcn_wmma_f32_16x16x32_bf16(false, (A), false, (B), (short)0, (C), false, false)

extern "C" __global__ void lstm_init_kernel(const float* __restrict__ hx,
                                            __bf16* __restrict__ hbuf0,
                                            unsigned* __restrict__ cnt) {
  unsigned i = blockIdx.x * blockDim.x + threadIdx.x;
  if (i == 0) *cnt = 0u;
  if (i < BSZ * HSZ) hbuf0[i] = (__bf16)hx[i];
}

extern "C" __global__ __launch_bounds__(THREADS)
void lstm_persistent_kernel(const float* __restrict__ x,
                            const float* __restrict__ cx,
                            const float* __restrict__ W_ih,
                            const float* __restrict__ W_hh,
                            const float* __restrict__ b_ih,
                            const float* __restrict__ b_hh,
                            float* __restrict__ out,
                            unsigned* __restrict__ barrier_cnt,
                            __bf16* __restrict__ hbuf0,
                            __bf16* __restrict__ hbuf1) {
  extern __shared__ char smem[];
  __bf16* Wl    = (__bf16*)(smem + OFF_W);     // [NGATE][WSTRIDE] W_hh slice, bf16
  __bf16* Hl    = (__bf16*)(smem + OFF_H);     // [BSZ][WSTRIDE]  h staging, bf16
  float*  Gl    = (float* )(smem + OFF_H);     // [BSZ][GSTRIDE]  gates f32 (aliases Hl)
  __bf16* Xl    = (__bf16*)(smem + OFF_X);     // [BSZ][XSTRIDE]  x_t padded bf16
  __bf16* WIl   = (__bf16*)(smem + OFF_WIH);   // [NGATE][XSTRIDE] W_ih padded
  float*  BIASl = (float* )(smem + OFF_BIAS);  // [NGATE]
  float*  Cl    = (float* )(smem + OFF_C);     // [BSZ][NH_LOCAL] cell state f32
  float*  Xraw  = (float* )(smem + OFF_XRAW);  // [BSZ*ISZ] raw f32 x_t (async dest)

  const int tid = threadIdx.x;
  const int wg  = blockIdx.x;        // 0..15
  const int j0  = wg * NH_LOCAL;     // hidden slice base

  // ---------------- one-time staging ----------------
  {
    // W_hh slice -> LDS bf16: local row n maps to global row (n>>5)*H + j0 + (n&31)
    const int n  = tid >> 2;               // 0..127
    const int cb = (tid & 3) * 128;        // 128-col block
    const int grow = (n >> 5) * HSZ + j0 + (n & 31);
    const float* src = W_hh + (size_t)grow * HSZ + cb;
    __bf16* dst = Wl + n * WSTRIDE + cb;
    for (int k = 0; k < 128; k += 4) {
      f32x4 v = *(const f32x4*)(src + k);
      dst[k+0] = (__bf16)v.x; dst[k+1] = (__bf16)v.y;
      dst[k+2] = (__bf16)v.z; dst[k+3] = (__bf16)v.w;
    }
  }
  if (tid < NGATE) {
    const int n = tid;
    const int grow = (n >> 5) * HSZ + j0 + (n & 31);
    const float* src = W_ih + (size_t)grow * ISZ;
    __bf16* dst = WIl + n * XSTRIDE;
    for (int i = 0; i < ISZ; ++i) dst[i] = (__bf16)src[i];
    for (int i = ISZ; i < 32; ++i) dst[i] = (__bf16)0.f;
    BIASl[n] = b_ih[grow] + b_hh[grow];
  }
  for (int f = tid; f < BSZ * NH_LOCAL; f += THREADS) {
    const int b = f >> 5, j = f & 31;
    Cl[b * NH_LOCAL + j] = cx[b * HSZ + j0 + j];
  }
  __syncthreads();

  const int lane = tid & 31;
  const int wave = tid >> 5;            // 0..15
  const int mq = wave >> 2, nq = wave & 3;
  const int m0a = mq * 32, m0b = m0a + 16;   // 2 M-tiles (batch)
  const int n0a = nq * 32, n0b = n0a + 16;   // 2 N-tiles (gate cols)
  const int nlo = lane & 15;

  for (int t = 0; t < T_STEPS; ++t) {
    const __bf16* hsrc = (t & 1) ? hbuf1 : hbuf0;
    __bf16*       hdst = (t & 1) ? hbuf0 : hbuf1;

    // ---- async stage h[B,H] (bf16) global -> padded LDS: 256B/thread, 16 x b128 ----
    {
      const int r  = tid >> 2;
      const int cb = (tid & 3) * 128;
      const unsigned dlo = lds_off(Hl + r * WSTRIDE + cb);
      const unsigned long long ga =
          (unsigned long long)(uintptr_t)(hsrc + r * HSZ + cb);
      #pragma unroll
      for (int k = 0; k < 16; ++k)
        asm volatile("global_load_async_to_lds_b128 %0, %1, off offset:%2"
                     :: "v"(dlo), "v"(ga), "i"(k * 16) : "memory");
    }
    // ---- async stage raw x_t (contiguous 5120B) global -> LDS ----
    if (tid < (BSZ * ISZ * 4) / 16) {  // 320 lanes x 16B
      const unsigned dlo = lds_off(Xraw) + tid * 16;
      const unsigned long long ga =
          (unsigned long long)(uintptr_t)(x + (size_t)t * BSZ * ISZ) + tid * 16;
      async_copy_b128(dlo, ga);
    }
    // warm L2 for next step's x block while we wait / compute
    if (t + 1 < T_STEPS && tid < BSZ)
      __builtin_prefetch(x + ((size_t)(t + 1) * BSZ + tid) * ISZ, 0, 1);

    wait_asynccnt0();
    __syncthreads();

    // convert raw x_t -> padded bf16 (K padded to 32)
    if (tid < BSZ) {
      const float* src = Xraw + tid * ISZ;
      __bf16* dst = Xl + tid * XSTRIDE;
      #pragma unroll
      for (int i = 0; i < ISZ; ++i) dst[i] = (__bf16)src[i];
      #pragma unroll
      for (int i = ISZ; i < 32; ++i) dst[i] = (__bf16)0.f;
    }
    __syncthreads();

    // ----- gates = bias + x_t@W_ih^T + h@W_hh^T  (WMMA bf16, f32 acc) -----
    const float ba = BIASl[n0a + nlo];
    const float bb = BIASl[n0b + nlo];
    v8f acc00, acc01, acc10, acc11;
    #pragma unroll
    for (int e = 0; e < 8; ++e) { acc00[e] = ba; acc01[e] = bb; acc10[e] = ba; acc11[e] = bb; }

    {  // x contribution: single K=32 WMMA per tile (cols 10..31 zero-padded)
      v16bf ax0 = load_frag(Xl,  m0a, XSTRIDE, 0, lane);
      v16bf ax1 = load_frag(Xl,  m0b, XSTRIDE, 0, lane);
      v16bf bw0 = load_frag(WIl, n0a, XSTRIDE, 0, lane);
      v16bf bw1 = load_frag(WIl, n0b, XSTRIDE, 0, lane);
      acc00 = WMMA_BF16(ax0, bw0, acc00);
      acc01 = WMMA_BF16(ax0, bw1, acc01);
      acc10 = WMMA_BF16(ax1, bw0, acc10);
      acc11 = WMMA_BF16(ax1, bw1, acc11);
    }
    #pragma unroll 2
    for (int k0 = 0; k0 < HSZ; k0 += 32) {   // recurrent GEMM, K=512
      v16bf a0 = load_frag(Hl, m0a, WSTRIDE, k0, lane);
      v16bf a1 = load_frag(Hl, m0b, WSTRIDE, k0, lane);
      v16bf b0 = load_frag(Wl, n0a, WSTRIDE, k0, lane);
      v16bf b1 = load_frag(Wl, n0b, WSTRIDE, k0, lane);
      acc00 = WMMA_BF16(a0, b0, acc00);
      acc01 = WMMA_BF16(a0, b1, acc01);
      acc10 = WMMA_BF16(a1, b0, acc10);
      acc11 = WMMA_BF16(a1, b1, acc11);
    }
    __syncthreads();   // all waves done reading Hl before gate buffer aliases it

    // scatter C fragments (lane L<16: M=v, L>=16: M=8+v; N=lane&15)
    {
      const int half = lane >> 4;
      #pragma unroll
      for (int v = 0; v < 8; ++v) {
        const int mA = m0a + v + 8 * half;
        const int mB = m0b + v + 8 * half;
        Gl[mA * GSTRIDE + n0a + nlo] = acc00[v];
        Gl[mA * GSTRIDE + n0b + nlo] = acc01[v];
        Gl[mB * GSTRIDE + n0a + nlo] = acc10[v];
        Gl[mB * GSTRIDE + n0b + nlo] = acc11[v];
      }
    }
    __syncthreads();

    // ----- elementwise LSTM update for owned slice (i,f,g,o at cols j,32+j,64+j,96+j) -----
    for (int f = tid; f < BSZ * NH_LOCAL; f += THREADS) {
      const int b = f >> 5, j = f & 31;
      const float gi = Gl[b * GSTRIDE + j];
      const float gf = Gl[b * GSTRIDE + 32 + j];
      const float gg = Gl[b * GSTRIDE + 64 + j];
      const float go = Gl[b * GSTRIDE + 96 + j];
      const float ig = 1.f / (1.f + __expf(-gi));
      const float fg = 1.f / (1.f + __expf(-gf));
      const float gt = 2.f / (1.f + __expf(-2.f * gg)) - 1.f;
      const float og = 1.f / (1.f + __expf(-go));
      const float c  = fg * Cl[b * NH_LOCAL + j] + ig * gt;
      const float h  = og * (2.f / (1.f + __expf(-2.f * c)) - 1.f);
      Cl[b * NH_LOCAL + j] = c;
      hdst[b * HSZ + j0 + j] = (__bf16)h;
      if (t == T_STEPS - 1) {
        out[                 b * HSZ + j0 + j] = h;   // return value h
        out[    BSZ * HSZ +  b * HSZ + j0 + j] = h;   // tuple hx
        out[2 * BSZ * HSZ +  b * HSZ + j0 + j] = c;   // tuple cx
      }
    }

    // ----- device-wide barrier (monotonic counter, 16 WGs, all resident) -----
    __threadfence();
    __syncthreads();
    if (tid == 0) {
      __hip_atomic_fetch_add(barrier_cnt, 1u, __ATOMIC_RELEASE, __HIP_MEMORY_SCOPE_AGENT);
      const unsigned target = (unsigned)(t + 1) * NWG;
      while (__hip_atomic_load(barrier_cnt, __ATOMIC_ACQUIRE, __HIP_MEMORY_SCOPE_AGENT) < target)
        __builtin_amdgcn_s_sleep(1);
    }
    __syncthreads();
    __threadfence();
  }
}

extern "C" void kernel_launch(void* const* d_in, const int* in_sizes, int n_in,
                              void* d_out, int out_size, void* d_ws, size_t ws_size,
                              hipStream_t stream) {
  (void)in_sizes; (void)n_in; (void)out_size; (void)ws_size;
  const float* x    = (const float*)d_in[0];
  const float* hx   = (const float*)d_in[1];
  const float* cx   = (const float*)d_in[2];
  const float* W_ih = (const float*)d_in[3];
  const float* W_hh = (const float*)d_in[4];
  const float* b_ih = (const float*)d_in[5];
  const float* b_hh = (const float*)d_in[6];
  float* out = (float*)d_out;

  char* ws = (char*)d_ws;
  unsigned* cnt = (unsigned*)ws;                              // barrier counter
  __bf16* hbuf0 = (__bf16*)(ws + 256);                        // ping
  __bf16* hbuf1 = (__bf16*)(ws + 256 + (size_t)BSZ * HSZ * 2);// pong

  lstm_init_kernel<<<(BSZ * HSZ + 255) / 256, 256, 0, stream>>>(hx, hbuf0, cnt);
  lstm_persistent_kernel<<<NWG, THREADS, LDS_BYTES, stream>>>(
      x, cx, W_ih, W_hh, b_ih, b_hh, out, cnt, hbuf0, hbuf1);
}
